// MultiheadAttention_33234456936510
// MI455X (gfx1250) — matrix operand order repaired and checked by reading, not compiled, not run on hardware
//
#include <hip/hip_runtime.h>
#include <hip/hip_bf16.h>
#include <cmath>
#include <stdint.h>

typedef __attribute__((ext_vector_type(16))) __bf16 v16bf;
typedef __attribute__((ext_vector_type(8)))  float  v8f;
typedef __attribute__((ext_vector_type(4)))  unsigned int u32x4;
typedef __attribute__((ext_vector_type(8)))  int    i32x8;
typedef __attribute__((ext_vector_type(4)))  int    i32x4;
typedef unsigned short u16;
typedef unsigned int   u32;

#define SEQ_T 1024
#define BSZ_  8
#define EMB   512
#define NH    8
#define HD    64
#define NEG_MAX (-3.402823466e38f)

// ---------- bf16 helpers (bit-level, round-to-nearest-even) ----------
__device__ __forceinline__ u16 f2bu(float f) {
  union { float f; u32 u; } x; x.f = f;
  u32 r = x.u + 0x7FFFu + ((x.u >> 16) & 1u);
  return (u16)(r >> 16);
}
__device__ __forceinline__ float bu2f(u16 u) {
  union { u32 u; float f; } x; x.u = ((u32)u) << 16; return x.f;
}
__device__ __forceinline__ __bf16 u2b(u16 u) {
  union { u16 u; __bf16 b; } x; x.u = u; return x.b;
}

__device__ __forceinline__ v8f wmma_bf16(v16bf a, v16bf b, v8f c) {
  // D = A(16x32) * B(32x16) + C, f32 accumulate
  return __builtin_amdgcn_wmma_f32_16x16x32_bf16(false, a, false, b, (short)0, c, false, false);
}

// ---------- Tensor Data Mover: 2-D bf16 tile -> padded LDS image ----------
// tile_d0 elements per row (= tensor_dim0 = tile_dim0), tile_d1 rows,
// row_stride in elements, pad code c inserts 1 DWORD pad every 2^(c+1) DWORDs.
__device__ __forceinline__ void tdm_load_2d(const u16* gsrc, unsigned lds_off,
                                            int tile_d0, int tile_d1,
                                            int row_stride, int pad_code)
{
  unsigned long long ga = (unsigned long long)(uintptr_t)gsrc;
  u32x4 g0;
  g0[0] = 1u;                                        // count=1 (valid user D#)
  g0[1] = lds_off;                                   // LDS byte address
  g0[2] = (unsigned)(ga & 0xFFFFFFFFu);              // global_addr[31:0]
  g0[3] = (unsigned)((ga >> 32) & 0x01FFFFFFu)       // global_addr[56:32]
          | (2u << 30);                              // type = 2 ("image")
  i32x8 g1;
  g1[0] = (1 << 16)                                  // data_size = 2 bytes
        | (1 << 20)                                  // pad_enable
        | (pad_code << 22);                          // pad_interval
  g1[1] = (tile_d0 << 16);                           // tensor_dim0[15:0] @ bits 63:48
  g1[2] = (tile_d1 << 16);                           // tensor_dim1[15:0] @ bits 95:80
  g1[3] = (tile_d0 << 16);                           // tile_dim0 @ bits 127:112
  g1[4] = tile_d1;                                   // tile_dim1 @ bits 143:128
  g1[5] = row_stride;                                // tensor_dim0_stride[31:0]
  g1[6] = 0;
  g1[7] = 0;
  i32x4 z4 = {0, 0, 0, 0};
#if __clang_major__ >= 23
  i32x8 z8 = {0, 0, 0, 0, 0, 0, 0, 0};
  __builtin_amdgcn_tensor_load_to_lds(g0, g1, z4, z4, z8, 0);
#else
  __builtin_amdgcn_tensor_load_to_lds(g0, g1, z4, z4, 0);
#endif
}

// ---------- fragment loaders from LDS (contiguous k order k = 16*half + j, same for A and B) ----
// A 16x32 bf16: lanes 0-15 row=lane, K 0..15; lanes 16-31 row=lane-16, K 16..31
__device__ __forceinline__ v16bf frag_a(const u16* lds, int ld, int row0, int k0, int lane) {
  int r = row0 + (lane & 15);
  int kb = k0 + ((lane & 16) ? 16 : 0);
  v16bf f;
#pragma unroll
  for (int j = 0; j < 16; ++j)
    f[j] = u2b(lds[r * ld + kb + j]);
  return f;
}
// B 32x16 bf16 from LDS stored [n][k]: lanes 0-15 col=lane K 0..15; 16-31 K 16..31
__device__ __forceinline__ v16bf frag_b_nk(const u16* lds, int ld, int n0, int k0, int lane) {
  int n = n0 + (lane & 15);
  int kb = k0 + ((lane & 16) ? 16 : 0);
  v16bf f;
#pragma unroll
  for (int j = 0; j < 16; ++j)
    f[j] = u2b(lds[n * ld + kb + j]);
  return f;
}
// B 32x16 bf16 from LDS stored [k][n] (e.g. V tile [s][d])
__device__ __forceinline__ v16bf frag_b_kn(const u16* lds, int ld, int k0, int n0, int lane) {
  int n = n0 + (lane & 15);
  int kb = k0 + ((lane & 16) ? 16 : 0);
  v16bf f;
#pragma unroll
  for (int j = 0; j < 16; ++j)
    f[j] = u2b(lds[(kb + j) * ld + n]);
  return f;
}

// ---------------- weight fp32 -> bf16 ----------------
__global__ __launch_bounds__(256) void cvt_weights_kernel(
    const float* __restrict__ Wi, const float* __restrict__ Wo,
    u16* __restrict__ Wib, u16* __restrict__ Wob)
{
  int gid = blockIdx.x * 256 + threadIdx.x;
  if (gid < 3 * EMB * EMB) Wib[gid] = f2bu(Wi[gid]);
  int g2 = gid - 3 * EMB * EMB;
  if (g2 >= 0 && g2 < EMB * EMB) Wob[g2] = f2bu(Wo[g2]);
}

// ---------------- in-projection GEMM: C[m,n] = X[m,:] . W[n,:] + b[n] -------
// X fp32 [8192,512] (converted on the fly), W bf16 via TDM, out bf16 head layout
__global__ __launch_bounds__(256) void proj_gemm_kernel(
    const float* __restrict__ X, const u16* __restrict__ W,
    const float* __restrict__ bias, u16* __restrict__ OutHead)
{
  __shared__ u16 As[128 * 34];
  __shared__ u16 Bs[64 * 34];
  const int tid = threadIdx.x;
  const int lane = tid & 31, wid = tid >> 5;
  const int ln = lane & 15, half = (lane >> 4) & 1;
  const int wm = wid >> 1, wn = wid & 1;
  const int mBase = blockIdx.y * 128, nBase = blockIdx.x * 64;

  v8f c00 = {}, c01 = {}, c10 = {}, c11 = {};
  const int arow = tid >> 1, akoff = (tid & 1) * 16;
  const unsigned bsOff = (unsigned)(uintptr_t)Bs;

  for (int k0 = 0; k0 < EMB; k0 += 32) {
    __syncthreads();                        // previous-tile consumers done
    if (wid == 0)                           // TDM: 64x32 bf16 weight tile
      tdm_load_2d(W + nBase * EMB + k0, bsOff, 32, 64, EMB, 3);
    if (k0 + 32 < EMB)
      __builtin_prefetch(&X[(mBase + arow) * EMB + k0 + 32 + akoff], 0, 1);
#pragma unroll
    for (int i = 0; i < 16; ++i)            // fp32 -> bf16 A staging
      As[arow * 34 + akoff + i] = f2bu(X[(mBase + arow) * EMB + k0 + akoff + i]);
    __builtin_amdgcn_s_wait_tensorcnt(0);
    __syncthreads();
    v16bf a0 = frag_a(As, 34, wm * 32,      0, lane);
    v16bf a1 = frag_a(As, 34, wm * 32 + 16, 0, lane);
    v16bf b0 = frag_b_nk(Bs, 34, wn * 32,      0, lane);
    v16bf b1 = frag_b_nk(Bs, 34, wn * 32 + 16, 0, lane);
    c00 = wmma_bf16(a0, b0, c00);
    c01 = wmma_bf16(a0, b1, c01);
    c10 = wmma_bf16(a1, b0, c10);
    c11 = wmma_bf16(a1, b1, c11);
  }

  v8f cc[2][2] = {{c00, c01}, {c10, c11}};
#pragma unroll
  for (int mi = 0; mi < 2; ++mi)
#pragma unroll
    for (int ni = 0; ni < 2; ++ni)
#pragma unroll
      for (int r = 0; r < 8; ++r) {
        int m = mBase + wm * 32 + mi * 16 + r + half * 8;
        int n = nBase + wn * 32 + ni * 16 + ln;
        float val = cc[mi][ni][r] + bias[n];
        int t = m >> 3, b = m & 7, h = n >> 6, d = n & 63;
        OutHead[((b * NH + h) * SEQ_T + t) * HD + d] = f2bu(val);
      }
}

// ---------------- out-projection GEMM: bf16 A & B via TDM, fp32 result ------
__global__ __launch_bounds__(256) void outproj_gemm_kernel(
    const u16* __restrict__ X, const u16* __restrict__ W,
    const float* __restrict__ bias, float* __restrict__ Out)
{
  __shared__ u16 As[128 * 34];
  __shared__ u16 Bs[64 * 34];
  const int tid = threadIdx.x;
  const int lane = tid & 31, wid = tid >> 5;
  const int ln = lane & 15, half = (lane >> 4) & 1;
  const int wm = wid >> 1, wn = wid & 1;
  const int mBase = blockIdx.y * 128, nBase = blockIdx.x * 64;

  v8f c00 = {}, c01 = {}, c10 = {}, c11 = {};
  const unsigned asOff = (unsigned)(uintptr_t)As;
  const unsigned bsOff = (unsigned)(uintptr_t)Bs;

  for (int k0 = 0; k0 < EMB; k0 += 32) {
    __syncthreads();
    if (wid == 0) {
      tdm_load_2d(X + mBase * EMB + k0, asOff, 32, 128, EMB, 3);
      tdm_load_2d(W + nBase * EMB + k0, bsOff, 32, 64,  EMB, 3);
    }
    __builtin_amdgcn_s_wait_tensorcnt(0);
    __syncthreads();
    v16bf a0 = frag_a(As, 34, wm * 32,      0, lane);
    v16bf a1 = frag_a(As, 34, wm * 32 + 16, 0, lane);
    v16bf b0 = frag_b_nk(Bs, 34, wn * 32,      0, lane);
    v16bf b1 = frag_b_nk(Bs, 34, wn * 32 + 16, 0, lane);
    c00 = wmma_bf16(a0, b0, c00);
    c01 = wmma_bf16(a0, b1, c01);
    c10 = wmma_bf16(a1, b0, c10);
    c11 = wmma_bf16(a1, b1, c11);
  }

  v8f cc[2][2] = {{c00, c01}, {c10, c11}};
#pragma unroll
  for (int mi = 0; mi < 2; ++mi)
#pragma unroll
    for (int ni = 0; ni < 2; ++ni)
#pragma unroll
      for (int r = 0; r < 8; ++r) {
        int m = mBase + wm * 32 + mi * 16 + r + half * 8;
        int n = nBase + wn * 32 + ni * 16 + ln;
        Out[m * EMB + n] = cc[mi][ni][r] + bias[n]; // (T1+T2)=1.0 folded
      }
}

// ---------------- fused SCA flash attention + SSA argmax ----------------
// grid (64 head-batches, 8 row-blocks of 128), 256 threads = 8 waves x 16 rows
__global__ __launch_bounds__(256) void flash_dual_kernel(
    const u16* __restrict__ Qc, const u16* __restrict__ Kc,
    const u16* __restrict__ Qs, const u16* __restrict__ Ks,
    const u16* __restrict__ Vh, const float* __restrict__ sem,
    float* __restrict__ Osca, int* __restrict__ IdxOut)
{
  __shared__ u16 KcL[64 * 66];
  __shared__ u16 KsL[64 * 66];
  __shared__ u16 VL [64 * 66];
  __shared__ u16 PL [8 * 16 * 68];

  const int tid = threadIdx.x;
  const int lane = tid & 31, wid = tid >> 5;
  const int ln = lane & 15, half = (lane >> 4) & 1;
  const int hb = blockIdx.x;
  const int rowBase = blockIdx.y * 128 + wid * 16;

  // Q fragments in registers, scale 1/sqrt(64)=0.125 folded (exact in bf16)
  v16bf qc[2], qs[2];
  {
    const u16* qcg = Qc + (size_t)(hb * SEQ_T + rowBase) * HD;
    const u16* qsg = Qs + (size_t)(hb * SEQ_T + rowBase) * HD;
#pragma unroll
    for (int kk = 0; kk < 2; ++kk)
#pragma unroll
      for (int j = 0; j < 16; ++j) {
        int k = kk * 32 + (half ? 16 : 0) + j;
        qc[kk][j] = u2b(f2bu(bu2f(qcg[ln * HD + k]) * 0.125f));
        qs[kk][j] = u2b(f2bu(bu2f(qsg[ln * HD + k]) * 0.125f));
      }
  }

  float mrow[8], lrow[8], bv[8];
  int bix[8];
#pragma unroll
  for (int r = 0; r < 8; ++r) { mrow[r] = -INFINITY; lrow[r] = 0.f; bv[r] = -INFINITY; bix[r] = 0; }
  v8f acc0 = {}, acc1 = {}, acc2 = {}, acc3 = {};
  u16* PW = PL + wid * 16 * 68;
  const unsigned kcOff = (unsigned)(uintptr_t)KcL;
  const unsigned ksOff = (unsigned)(uintptr_t)KsL;
  const unsigned vOff  = (unsigned)(uintptr_t)VL;

  for (int st = 0; st < 16; ++st) {
    __syncthreads();                       // everyone done with previous tiles
    if (wid == 0) {                        // TDM streams the 3 64x64 bf16 tiles
      const size_t tb = (size_t)(hb * SEQ_T + st * 64) * HD;
      tdm_load_2d(Kc + tb, kcOff, HD, 64, HD, 4);
      tdm_load_2d(Ks + tb, ksOff, HD, 64, HD, 4);
      tdm_load_2d(Vh + tb, vOff,  HD, 64, HD, 4);
    }
    __builtin_amdgcn_s_wait_tensorcnt(0);
    __syncthreads();

    v8f ssc[4];
#pragma unroll
    for (int j = 0; j < 4; ++j) {
      v8f cs = {}, cx = {};
#pragma unroll
      for (int kk = 0; kk < 2; ++kk) {
        cs = wmma_bf16(qc[kk], frag_b_nk(KcL, 66, j * 16, kk * 32, lane), cs);
        cx = wmma_bf16(qs[kk], frag_b_nk(KsL, 66, j * 16, kk * 32, lane), cx);
      }
      int s = st * 64 + j * 16 + ln;
#pragma unroll
      for (int r = 0; r < 8; ++r) {
        int t = rowBase + r + half * 8;
        bool keep = sem[t * SEQ_T + s] >= 0.5f;
        float v1 = keep ? cs[r] : NEG_MAX;
        float v2 = keep ? cx[r] : NEG_MAX;
        cs[r] = v1;
        if (v2 > bv[r] || (v2 == bv[r] && s < bix[r])) { bv[r] = v2; bix[r] = s; }
      }
      ssc[j] = cs;
    }

    // online softmax (row reduction within each 16-lane half = C-frag rows)
#pragma unroll
    for (int r = 0; r < 8; ++r) {
      float tm = fmaxf(fmaxf(ssc[0][r], ssc[1][r]), fmaxf(ssc[2][r], ssc[3][r]));
#pragma unroll
      for (int mk = 1; mk <= 8; mk <<= 1) tm = fmaxf(tm, __shfl_xor(tm, mk));
      float mn = fmaxf(mrow[r], tm);
      float es = __expf(mrow[r] - mn);
      float rs = 0.f;
#pragma unroll
      for (int j = 0; j < 4; ++j) { float pp = __expf(ssc[j][r] - mn); ssc[j][r] = pp; rs += pp; }
#pragma unroll
      for (int mk = 1; mk <= 8; mk <<= 1) rs += __shfl_xor(rs, mk);
      lrow[r] = lrow[r] * es + rs;
      mrow[r] = mn;
      acc0[r] *= es; acc1[r] *= es; acc2[r] *= es; acc3[r] *= es;
    }

    // re-lay-out P (C-frag -> A-frag) through per-wave LDS tile
#pragma unroll
    for (int j = 0; j < 4; ++j)
#pragma unroll
      for (int r = 0; r < 8; ++r)
        PW[(r + half * 8) * 68 + j * 16 + ln] = f2bu(ssc[j][r]);
    __syncthreads();

    // acc += P(16x64) @ V(64x64)
#pragma unroll
    for (int sk = 0; sk < 2; ++sk) {
      v16bf pa = frag_a(PW, 68, 0, sk * 32, lane);
      acc0 = wmma_bf16(pa, frag_b_kn(VL, 66, sk * 32,  0, lane), acc0);
      acc1 = wmma_bf16(pa, frag_b_kn(VL, 66, sk * 32, 16, lane), acc1);
      acc2 = wmma_bf16(pa, frag_b_kn(VL, 66, sk * 32, 32, lane), acc2);
      acc3 = wmma_bf16(pa, frag_b_kn(VL, 66, sk * 32, 48, lane), acc3);
    }
  }

  // epilogue: normalize + store SCA output fp32 (merged-head layout)
  const int b = hb >> 3, h = hb & 7;
  v8f accs[4] = {acc0, acc1, acc2, acc3};
#pragma unroll
  for (int dt = 0; dt < 4; ++dt)
#pragma unroll
    for (int r = 0; r < 8; ++r) {
      int t = rowBase + r + half * 8;
      int d = dt * 16 + ln;
      Osca[(t * BSZ_ + b) * EMB + h * HD + d] = accs[dt][r] / lrow[r];
    }

  // SSA argmax reduction + store
#pragma unroll
  for (int r = 0; r < 8; ++r) {
    float v = bv[r]; int ix = bix[r];
#pragma unroll
    for (int mk = 1; mk <= 8; mk <<= 1) {
      float ov = __shfl_xor(v, mk);
      int oi = __shfl_xor(ix, mk);
      if (ov > v || (ov == v && oi < ix)) { v = ov; ix = oi; }
    }
    if (ln == 0) IdxOut[hb * SEQ_T + rowBase + r + half * 8] = ix;
  }
}

// ------------- combine: P = bf16(T1*O_sca + T2*V[argmax]) -------------
__global__ __launch_bounds__(256) void combine_kernel(
    const float* __restrict__ Osca, const u16* __restrict__ Vh,
    const int* __restrict__ Idx, u16* __restrict__ Pmat)
{
  int gid = blockIdx.x * 256 + threadIdx.x;
  int m = gid >> 9, n = gid & 511;
  int t = m >> 3, b = m & 7, h = n >> 6, d = n & 63;
  int hb = b * NH + h;
  int j = Idx[hb * SEQ_T + t];
  float v = bu2f(Vh[(hb * SEQ_T + j) * HD + d]);
  Pmat[gid] = f2bu(0.6f * Osca[gid] + 0.4f * v);
}

// --------------------------------------------------------------------------
extern "C" void kernel_launch(void* const* d_in, const int* in_sizes, int n_in,
                              void* d_out, int out_size, void* d_ws, size_t ws_size,
                              hipStream_t stream) {
  const float* query     = (const float*)d_in[0];
  const float* key_      = (const float*)d_in[1];
  const float* value     = (const float*)d_in[2];
  const float* query_sem = (const float*)d_in[3];
  const float* key_sem   = (const float*)d_in[4];
  const float* sem_map   = (const float*)d_in[5];
  const float* Wi        = (const float*)d_in[6];
  const float* bi        = (const float*)d_in[7];
  const float* Wo        = (const float*)d_in[8];
  const float* bo        = (const float*)d_in[9];
  float* out = (float*)d_out;

  char* p = (char*)d_ws;
  u16* Wib = (u16*)p;  p += (size_t)3 * EMB * EMB * 2;
  u16* Wob = (u16*)p;  p += (size_t)EMB * EMB * 2;
  const size_t headBytes = (size_t)64 * SEQ_T * HD * 2;
  u16* Qc = (u16*)p; p += headBytes;
  u16* Kc = (u16*)p; p += headBytes;
  u16* Vh = (u16*)p; p += headBytes;
  u16* Qs = (u16*)p; p += headBytes;
  u16* Ks = (u16*)p; p += headBytes;
  float* Osca = (float*)p; p += (size_t)8192 * EMB * 4;
  int*   Idx  = (int*)p;   p += (size_t)64 * SEQ_T * 4;
  u16*   Pmat = (u16*)p;   p += (size_t)8192 * EMB * 2;

  hipLaunchKernelGGL(cvt_weights_kernel, dim3(4096), dim3(256), 0, stream, Wi, Wo, Wib, Wob);

  dim3 g(8, 64);
  hipLaunchKernelGGL(proj_gemm_kernel, g, dim3(256), 0, stream, query_sem, Wib,                 bi,        Qc);
  hipLaunchKernelGGL(proj_gemm_kernel, g, dim3(256), 0, stream, key_sem,   Wib + 512 * 512,     bi + 512,  Kc);
  hipLaunchKernelGGL(proj_gemm_kernel, g, dim3(256), 0, stream, value,     Wib + 2 * 512 * 512, bi + 1024, Vh);
  hipLaunchKernelGGL(proj_gemm_kernel, g, dim3(256), 0, stream, query,     Wib,                 bi,        Qs);
  hipLaunchKernelGGL(proj_gemm_kernel, g, dim3(256), 0, stream, key_,      Wib + 512 * 512,     bi + 512,  Ks);

  hipLaunchKernelGGL(flash_dual_kernel, dim3(64, 8), dim3(256), 0, stream,
                     Qc, Kc, Qs, Ks, Vh, sem_map, Osca, Idx);
  hipLaunchKernelGGL(combine_kernel, dim3(16384), dim3(256), 0, stream, Osca, Vh, Idx, Pmat);
  hipLaunchKernelGGL(outproj_gemm_kernel, g, dim3(256), 0, stream, Pmat, Wob, bo, out);
}
